// Fused_27109833572751
// MI455X (gfx1250) — compile-verified, matrix operand order checked by reading, not run on hardware
//
#include <hip/hip_runtime.h>
#include <math.h>

typedef __bf16 bf16_t;
typedef __attribute__((ext_vector_type(16))) __bf16 v16bf;
typedef __attribute__((ext_vector_type(8)))  float  v8f;

#define NTHREADS 256

// ---------------------------------------------------------------------------
// WMMA helpers
// ---------------------------------------------------------------------------
__device__ __forceinline__ v8f wmma_bf16(v16bf a, v16bf b, v8f c) {
  // D = A(16x32) * B(32x16) + C, f32 accumulate
  return __builtin_amdgcn_wmma_f32_16x16x32_bf16(false, a, false, b, (short)0, c,
                                                 false, false);
}

// Load one 16x32 bf16 fragment (A or symmetric-B) from a row-major LDS tile.
// CDNA5 layout (ISA 7.12.2): lanes 0-15 -> rows, K runs {0..7, 16..23};
// lanes 16-31 -> K runs {8..15, 24..31}. Two 16B ds loads per fragment.
__device__ __forceinline__ v16bf ld_frag(const bf16_t* tile, int ld, int row,
                                         int kbase) {
  const int lane = threadIdx.x & 31;
  const int hi   = lane >> 4;
  const bf16_t* p = tile + (size_t)(row + (lane & 15)) * ld + kbase + hi * 8;
  v16bf f;
  ((uint4*)&f)[0] = *(const uint4*)(p);
  ((uint4*)&f)[1] = *(const uint4*)(p + 16);
  return f;
}

// ---------------------------------------------------------------------------
// 1) RMSNorm: writes xn = rmsnorm(x)*rms_w as bf16, and xb = bf16(x)
// ---------------------------------------------------------------------------
__global__ __launch_bounds__(NTHREADS) void rms_kernel(const float* __restrict__ x,
                                                       const float* __restrict__ rms_w,
                                                       bf16_t* __restrict__ xn,
                                                       bf16_t* __restrict__ xb) {
  __shared__ float red[NTHREADS];
  const int C = 1024;
  const int bt = blockIdx.x;
  const int tid = threadIdx.x;
  const size_t off = (size_t)bt * C;
  float vals[4];
  float ss = 0.f;
#pragma unroll
  for (int i = 0; i < 4; ++i) {
    vals[i] = x[off + tid + i * NTHREADS];
    ss += vals[i] * vals[i];
  }
  red[tid] = ss;
  __syncthreads();
  for (int s = NTHREADS / 2; s > 0; s >>= 1) {
    if (tid < s) red[tid] += red[tid + s];
    __syncthreads();
  }
  const float nrm = sqrtf(red[0]) * 0.03125f; // * C^-0.5 (1/32)
  const float inv = 1.0f / fmaxf(nrm, 1e-8f);
#pragma unroll
  for (int i = 0; i < 4; ++i) {
    const int c = tid + i * NTHREADS;
    xn[off + c] = (bf16_t)(vals[i] * inv * rms_w[c]);
    xb[off + c] = (bf16_t)vals[i];
  }
}

// ---------------------------------------------------------------------------
// 2) fp32 -> bf16 convert (weights)
// ---------------------------------------------------------------------------
__global__ void cvt_bf16_kernel(const float* __restrict__ in,
                                bf16_t* __restrict__ out, int n) {
  int i = blockIdx.x * NTHREADS + threadIdx.x;
  if (i < n) out[i] = (bf16_t)in[i];
}

// ---------------------------------------------------------------------------
// 3) Tiled bf16 WMMA GEMM:  C[M,N] (f32) = A[M,K] * B[N,K]^T
//    BM=128, BN=128, BK=32; 8 waves (2 along M x 4 along N), wave = 64x32.
// ---------------------------------------------------------------------------
#define BM 128
#define BN 128
#define BK 32

__global__ __launch_bounds__(NTHREADS) void gemm_bf16_kernel(
    const bf16_t* __restrict__ A, const bf16_t* __restrict__ Bm,
    float* __restrict__ C, int M, int N, int K) {
  __shared__ alignas(16) bf16_t As[BM * BK];
  __shared__ alignas(16) bf16_t Bs[BN * BK];
  const int tid = threadIdx.x;
  const int wave = tid >> 5;
  const int lane = tid & 31;
  const int bm = blockIdx.y * BM;
  const int bn = blockIdx.x * BN;
  const int wm = (wave >> 2) * 64; // wave row offset in tile
  const int wn = (wave & 3) * 32;  // wave col offset in tile
  v8f acc[4][2] = {};

  const int lr = tid >> 1;          // row this thread loads (0..127)
  const int lc = (tid & 1) * 16;    // col offset (0 or 16)

  for (int k0 = 0; k0 < K; k0 += BK) {
    // cooperative tile loads: 16 bf16 per thread per matrix
    {
      const bf16_t* ga = A + (size_t)(bm + lr) * K + k0 + lc;
      const bf16_t* gb = Bm + (size_t)(bn + lr) * K + k0 + lc;
      const int idx = lr * BK + lc;
      *(uint4*)&As[idx]     = *(const uint4*)(ga);
      *(uint4*)&As[idx + 8] = *(const uint4*)(ga + 8);
      *(uint4*)&Bs[idx]     = *(const uint4*)(gb);
      *(uint4*)&Bs[idx + 8] = *(const uint4*)(gb + 8);
      if (k0 + BK < K) { // keep the HBM pipe warm (global_prefetch_b8)
        __builtin_prefetch(ga + BK, 0, 0);
        __builtin_prefetch(gb + BK, 0, 0);
      }
    }
    __syncthreads();
    v16bf bfrag[2];
#pragma unroll
    for (int j = 0; j < 2; ++j) bfrag[j] = ld_frag(Bs, BK, wn + j * 16, 0);
#pragma unroll
    for (int i = 0; i < 4; ++i) {
      v16bf afrag = ld_frag(As, BK, wm + i * 16, 0);
#pragma unroll
      for (int j = 0; j < 2; ++j)
        acc[i][j] = wmma_bf16(afrag, bfrag[j], acc[i][j]);
    }
    __syncthreads();
  }

  const int hi = lane >> 4;
  const int col0 = lane & 15;
#pragma unroll
  for (int i = 0; i < 4; ++i)
#pragma unroll
    for (int j = 0; j < 2; ++j)
#pragma unroll
      for (int e = 0; e < 8; ++e) {
        const int row = bm + wm + i * 16 + e + hi * 8;
        const int col = bn + wn + j * 16 + col0;
        C[(size_t)row * N + col] = acc[i][j][e];
      }
}

// ---------------------------------------------------------------------------
// 4) xPos rotary on q,k; scatter q,k,v to [B,H,T,64] bf16
// ---------------------------------------------------------------------------
__global__ __launch_bounds__(NTHREADS) void xpos_kernel(
    const float* __restrict__ qkvff, bf16_t* __restrict__ qg,
    bf16_t* __restrict__ kg, bf16_t* __restrict__ vg) {
  const int B = 2, T = 2048, C = 1024, H = 16;
  const int bt = blockIdx.x;
  const int b = bt / T, t = bt % T;
  const int tid = threadIdx.x;
  const size_t rowoff = (size_t)bt * (5 * C);
  const float power = ((float)t - 1024.0f) * (1.0f / 512.0f); // (t+min_pos)/SCALE_BASE
  const float LN1E4 = 9.210340371976184f;

  for (int i = tid; i < C / 2; i += NTHREADS) {
    const float sv = (2.0f * i + 0.4f * C) / (1.4f * C);
    const float scale = powf(sv, power);
    const float invf = expf(-LN1E4 * (float)i / 512.0f);
    const float ang = (float)t * invf;
    const float s = sinf(ang), c = cosf(ang);
    const int h = (2 * i) >> 6;
    const int d = (2 * i) & 63;
    const size_t o = (((size_t)b * H + h) * T + t) << 6;
    { // q: upscale
      const float sq = s * scale, cq = c * scale;
      const float x0 = qkvff[rowoff + 2 * i];
      const float x1 = qkvff[rowoff + 2 * i + 1];
      qg[o + d]     = (bf16_t)(x0 * cq - x1 * sq);
      qg[o + d + 1] = (bf16_t)(x1 * cq + x0 * sq);
    }
    { // k: downscale
      const float ks = 1.0f / scale;
      const float sk = s * ks, ck = c * ks;
      const float x0 = qkvff[rowoff + C + 2 * i];
      const float x1 = qkvff[rowoff + C + 2 * i + 1];
      kg[o + d]     = (bf16_t)(x0 * ck - x1 * sk);
      kg[o + d + 1] = (bf16_t)(x1 * ck + x0 * sk);
    }
  }
  for (int i = tid; i < C; i += NTHREADS) {
    const int h = i >> 6, d = i & 63;
    vg[((((size_t)b * H + h) * T + t) << 6) + d] =
        (bf16_t)qkvff[rowoff + 2 * C + i];
  }
}

// ---------------------------------------------------------------------------
// 5) exact GELU on ff slice -> bf16 [4096, 2048]
// ---------------------------------------------------------------------------
__global__ void gelu_kernel(const float* __restrict__ qkvff,
                            bf16_t* __restrict__ g) {
  const int n = 4096 * 2048;
  int i = blockIdx.x * NTHREADS + threadIdx.x;
  if (i >= n) return;
  const int bt = i >> 11;        // /2048
  const int e = i & 2047;
  const float x = qkvff[(size_t)bt * 5120 + 3072 + e];
  g[i] = (bf16_t)(0.5f * x * (1.0f + erff(x * 0.70710678118654752f)));
}

// ---------------------------------------------------------------------------
// 6) Retention attention: per (b,h, 64-row t block).
//    S = (Q K^T) * hd^-0.5 * gamma^(t-s) [causal], Y = S V.  All WMMA bf16.
// ---------------------------------------------------------------------------
__global__ __launch_bounds__(NTHREADS) void attn_kernel(
    const bf16_t* __restrict__ qg, const bf16_t* __restrict__ kg,
    const bf16_t* __restrict__ vg, float* __restrict__ yg) {
  const int T = 2048, H = 16;
  __shared__ alignas(16) bf16_t Qs[64 * 64];
  __shared__ alignas(16) bf16_t Ks[64 * 64];
  __shared__ alignas(16) bf16_t VTs[64 * 64]; // V transposed: [d][s]
  __shared__ alignas(16) bf16_t Ss[64 * 64];
  const int bh = blockIdx.y;
  const int h = bh % H;
  const int tb = blockIdx.x;
  const int tid = threadIdx.x;
  const int wave = tid >> 5, lane = tid & 31;
  const int mt = wave >> 1;       // m sub-tile (0..3)
  const int nb = (wave & 1) * 2;  // n tile pair base (0 or 2)
  const int hi = lane >> 4, ln = lane & 15;
  const float gamma = 1.0f - exp2f(-5.0f - (float)h);
  const float lgam = logf(gamma);
  const float sc = 0.125f; // 64^-0.5
  const size_t base = ((size_t)bh * T) << 6;

  { // Q tile (resident whole kernel)
    const int idx = tid * 16;
    const int r = idx >> 6, c = idx & 63;
    const bf16_t* g = qg + base + ((size_t)(tb * 64 + r) << 6) + c;
    *(uint4*)&Qs[idx]     = *(const uint4*)(g);
    *(uint4*)&Qs[idx + 8] = *(const uint4*)(g + 8);
  }

  v8f yacc[2] = {};
  for (int sb = 0; sb <= tb; ++sb) {
    __syncthreads(); // Q ready / previous iter consumers done
    { // load K tile and V tile (transposed into LDS)
      const int idx = tid * 16;
      const int r = idx >> 6, c = idx & 63;
      const bf16_t* gk = kg + base + ((size_t)(sb * 64 + r) << 6) + c;
      *(uint4*)&Ks[idx]     = *(const uint4*)(gk);
      *(uint4*)&Ks[idx + 8] = *(const uint4*)(gk + 8);
      const bf16_t* gv = vg + base + ((size_t)(sb * 64 + r) << 6) + c;
      bf16_t tmpv[16];
      *(uint4*)&tmpv[0] = *(const uint4*)(gv);
      *(uint4*)&tmpv[8] = *(const uint4*)(gv + 8);
#pragma unroll
      for (int e = 0; e < 16; ++e) VTs[(c + e) * 64 + r] = tmpv[e];
    }
    __syncthreads();

    // S = Q * K^T  (M=64, N=64, K=hd=64)
    v8f sacc[2] = {};
#pragma unroll
    for (int ks = 0; ks < 64; ks += 32) {
      v16bf aq = ld_frag(Qs, 64, mt * 16, ks);
#pragma unroll
      for (int j = 0; j < 2; ++j) {
        v16bf bk = ld_frag(Ks, 64, (nb + j) * 16, ks);
        sacc[j] = wmma_bf16(aq, bk, sacc[j]);
      }
    }
    // decay mask + scale, restage S as bf16 (row-major 64x64)
#pragma unroll
    for (int j = 0; j < 2; ++j) {
      const int scol = (nb + j) * 16 + ln;
      const int sgl = sb * 64 + scol;
#pragma unroll
      for (int e = 0; e < 8; ++e) {
        const int m = mt * 16 + e + hi * 8;
        const int tgl = tb * 64 + m;
        const float diff = (float)(tgl - sgl);
        const float mlt = (tgl >= sgl) ? sc * expf(lgam * diff) : 0.0f;
        Ss[m * 64 + scol] = (bf16_t)(sacc[j][e] * mlt);
      }
    }
    __syncthreads();

    // Y += S * V  (K = 64 along s)
#pragma unroll
    for (int ks = 0; ks < 64; ks += 32) {
      v16bf as = ld_frag(Ss, 64, mt * 16, ks);
#pragma unroll
      for (int j = 0; j < 2; ++j) {
        v16bf bv = ld_frag(VTs, 64, (nb + j) * 16, ks);
        yacc[j] = wmma_bf16(as, bv, yacc[j]);
      }
    }
  }

#pragma unroll
  for (int j = 0; j < 2; ++j) {
    const int d = (nb + j) * 16 + ln;
#pragma unroll
    for (int e = 0; e < 8; ++e) {
      const int tgl = tb * 64 + mt * 16 + e + hi * 8;
      yg[base + ((size_t)tgl << 6) + d] = yacc[j][e];
    }
  }
}

// ---------------------------------------------------------------------------
// 7) GroupNorm over head dim + SiLU gate -> bf16 [4096, 1024]
// ---------------------------------------------------------------------------
__global__ __launch_bounds__(NTHREADS) void gn_gate_kernel(
    const float* __restrict__ yg, const float* __restrict__ gated,
    const float* __restrict__ b_gated, const float* __restrict__ gn_w,
    const float* __restrict__ gn_b, bf16_t* __restrict__ ygn) {
  const int T = 2048, H = 16;
  __shared__ float r1[NTHREADS];
  __shared__ float r2[NTHREADS];
  const int tid = threadIdx.x;
  const int grp = tid >> 6;  // 4 groups of 64
  const int l = tid & 63;
  const long gi = (long)blockIdx.x * 4 + grp; // over B*T*H
  const int h = (int)(gi % H);
  const long bt = gi / H;
  const int t = (int)(bt % T);
  const long b = bt / T;
  const size_t yoff = (((size_t)b * H + h) * T + t) << 6;
  const float v = yg[yoff + l];
  r1[tid] = v;
  r2[tid] = v * v;
  __syncthreads();
  for (int s = 32; s > 0; s >>= 1) {
    if (l < s) { r1[tid] += r1[tid + s]; r2[tid] += r2[tid + s]; }
    __syncthreads();
  }
  const float mu = r1[grp * 64] * (1.0f / 64.0f);
  const float var = r2[grp * 64] * (1.0f / 64.0f) - mu * mu;
  const float yn = (v - mu) * rsqrtf(var + 1e-5f);
  const int c = h * 64 + l;
  const float val = yn * gn_w[c] + gn_b[c];
  const float g = gated[(size_t)bt * 1024 + c] + b_gated[c];
  const float si = g / (1.0f + expf(-g));
  ygn[(size_t)bt * 1024 + c] = (bf16_t)(si * val);
}

// ---------------------------------------------------------------------------
// 8) Final residual combine: out = x + ff_out + proj + b_proj
// ---------------------------------------------------------------------------
__global__ void final_kernel(const float* __restrict__ x,
                             const float* __restrict__ ffout,
                             const float* __restrict__ proj,
                             const float* __restrict__ b_proj,
                             float* __restrict__ out, int n) {
  int i = blockIdx.x * NTHREADS + threadIdx.x;
  if (i < n) out[i] = x[i] + ffout[i] + proj[i] + b_proj[i & 1023];
}

// ---------------------------------------------------------------------------
// launch
// ---------------------------------------------------------------------------
extern "C" void kernel_launch(void* const* d_in, const int* in_sizes, int n_in,
                              void* d_out, int out_size, void* d_ws,
                              size_t ws_size, hipStream_t stream) {
  const float* x       = (const float*)d_in[0];
  const float* w_qkvff = (const float*)d_in[1];
  const float* w_gated = (const float*)d_in[2];
  const float* b_gated = (const float*)d_in[3];
  const float* w_proj  = (const float*)d_in[4];
  const float* b_proj  = (const float*)d_in[5];
  const float* gn_w    = (const float*)d_in[6];
  const float* gn_b    = (const float*)d_in[7];
  const float* w_ff    = (const float*)d_in[8];
  const float* rms_w   = (const float*)d_in[9];
  float* out = (float*)d_out;

  const int B = 2, T = 2048, C = 1024, H = 16;
  const size_t BT = (size_t)B * T; // 4096

  // workspace layout (bytes)
  char* ws = (char*)d_ws;
  size_t off = 0;
  auto take = [&](size_t bytes) { char* p = ws + off; off += (bytes + 255) & ~(size_t)255; return p; };
  bf16_t* xn   = (bf16_t*)take(BT * C * 2);           // rmsnormed x, bf16
  bf16_t* xb   = (bf16_t*)take(BT * C * 2);           // raw x, bf16
  bf16_t* wq   = (bf16_t*)take((size_t)5 * C * C * 2);
  bf16_t* wg   = (bf16_t*)take((size_t)C * C * 2);
  bf16_t* wp   = (bf16_t*)take((size_t)C * C * 2);
  bf16_t* wf   = (bf16_t*)take((size_t)C * 2 * C * 2);
  float*  qkvf = (float*)take(BT * 5 * C * 4);        // 84 MB, recycled below
  bf16_t* qg   = (bf16_t*)take(BT * C * 2);
  bf16_t* kg   = (bf16_t*)take(BT * C * 2);
  bf16_t* vg   = (bf16_t*)take(BT * C * 2);
  bf16_t* gf   = (bf16_t*)take(BT * 2 * C * 2);       // gelu(ff)
  bf16_t* ygn  = (bf16_t*)take(BT * C * 2);
  // recycle qkvff region after rotary+gelu consume it (stream-ordered)
  float* gated = qkvf;                                 // 16.8 MB
  float* yatt  = (float*)((char*)qkvf + (size_t)16777216);
  float* proj  = (float*)((char*)qkvf + (size_t)2 * 16777216);
  float* ffout = (float*)((char*)qkvf + (size_t)3 * 16777216);

  // 1) rmsnorm + bf16 casts
  rms_kernel<<<(int)BT, NTHREADS, 0, stream>>>(x, rms_w, xn, xb);
  // 2) weight conversions
  {
    int n = 5 * C * C;
    cvt_bf16_kernel<<<(n + 255) / 256, 256, 0, stream>>>(w_qkvff, wq, n);
    n = C * C;
    cvt_bf16_kernel<<<(n + 255) / 256, 256, 0, stream>>>(w_gated, wg, n);
    cvt_bf16_kernel<<<(n + 255) / 256, 256, 0, stream>>>(w_proj, wp, n);
    n = C * 2 * C;
    cvt_bf16_kernel<<<(n + 255) / 256, 256, 0, stream>>>(w_ff, wf, n);
  }
  // 3) qkvff GEMM: [4096,1024] x [5120,1024]^T -> [4096,5120]
  gemm_bf16_kernel<<<dim3(5 * C / BN, BT / BM), NTHREADS, 0, stream>>>(
      xn, wq, qkvf, (int)BT, 5 * C, C);
  // 4) rotary + head scatter
  xpos_kernel<<<(int)BT, NTHREADS, 0, stream>>>(qkvf, qg, kg, vg);
  // 5) gelu on ff slice
  {
    int n = (int)BT * 2 * C;
    gelu_kernel<<<(n + 255) / 256, 256, 0, stream>>>(qkvf, gf);
  }
  // 6) gated GEMM (uses raw x): [4096,1024] x [1024,1024]^T
  gemm_bf16_kernel<<<dim3(C / BN, BT / BM), NTHREADS, 0, stream>>>(
      xb, wg, gated, (int)BT, C, C);
  // 7) retention attention
  attn_kernel<<<dim3(T / 64, B * H), NTHREADS, 0, stream>>>(qg, kg, vg, yatt);
  // 8) groupnorm + silu gate
  gn_gate_kernel<<<(int)(BT * H / 4), NTHREADS, 0, stream>>>(
      yatt, gated, b_gated, gn_w, gn_b, ygn);
  // 9) output projection: [4096,1024] x [1024,1024]^T
  gemm_bf16_kernel<<<dim3(C / BN, BT / BM), NTHREADS, 0, stream>>>(
      ygn, wp, proj, (int)BT, C, C);
  // 10) FFN second GEMM: [4096,2048] x [1024,2048]^T
  gemm_bf16_kernel<<<dim3(C / BN, BT / BM), NTHREADS, 0, stream>>>(
      gf, wf, ffout, (int)BT, C, 2 * C);
  // 11) residual combine
  {
    int n = (int)(BT * C);
    final_kernel<<<(n + 255) / 256, 256, 0, stream>>>(x, ffout, proj, b_proj,
                                                      out, n);
  }
}
